// LSHNNetwork_5531917877583
// MI455X (gfx1250) — matrix operand order repaired and checked by reading, not compiled, not run on hardware
//
#include <hip/hip_runtime.h>
#include <hip/hip_bf16.h>

// ---------------------------------------------------------------------------
// Types / helpers
// ---------------------------------------------------------------------------
typedef __bf16 bf16_t;
typedef __attribute__((ext_vector_type(16))) __bf16 v16bf;
typedef __attribute__((ext_vector_type(8)))  float  v8f;
typedef __attribute__((ext_vector_type(4)))  unsigned int v4u;

union V16U { v16bf v; v4u q[2]; };

__device__ __forceinline__ bf16_t f2bf(float f) {
  unsigned u = __builtin_bit_cast(unsigned, f);
  u += 0x7FFFu + ((u >> 16) & 1u);                 // round-to-nearest-even
  unsigned short h = (unsigned short)(u >> 16);
  return __builtin_bit_cast(bf16_t, h);
}
__device__ __forceinline__ float bf2f(bf16_t b) {
  unsigned short h = __builtin_bit_cast(unsigned short, b);
  unsigned u = ((unsigned)h) << 16;
  return __builtin_bit_cast(float, u);
}
__device__ __forceinline__ int dmin(int a, int b) { return a < b ? a : b; }

// CDNA5 async global->LDS copy (16B per lane), tracked by ASYNCcnt.
// vdst = per-lane LDS byte address (low 32 bits of generic shared pointer,
// per ISA 10.2 LDS-aperture rule), vaddr = 64-bit global address.
__device__ __forceinline__ void async_ld16(unsigned lds, const float* g) {
  asm volatile("global_load_async_to_lds_b128 %0, %1, off"
               :: "v"(lds), "v"((unsigned long long)(const void*)g)
               : "memory");
}
__device__ __forceinline__ void wait_async0() {
  asm volatile("s_wait_asynccnt 0x0" ::: "memory");
}
__device__ __forceinline__ unsigned lds_addr(const void* p) {
  return (unsigned)(unsigned long long)p;          // low 32 bits = LDS offset
}

#define BQ      64      // batch (query rows)
#define NHD     8       // heads
#define PDIM    512
#define NSPLIT  64      // M-splits for flash partials
#define CH      32      // mem rows staged per iteration
#define SIMS_CHUNK 1024
#define SCALE_F 0.8f

// Core wave-level bf16 WMMA K-loop.  A: row-major pointer already offset to
// the tile row for this lane (M = lane&15).  Wt: "transposed-B" storage, row n
// holds K contiguously, pointer already offset to row (ct*16+ml).
// g = lane>>4 selects the K-half per the CDNA5 16-bit A/B VGPR layouts.
__device__ __forceinline__ v8f bf16_mma_loop(const bf16_t* __restrict__ Arow,
                                             const bf16_t* __restrict__ Wrow,
                                             int K, int g, v8f acc) {
  for (int k0 = 0; k0 < K; k0 += 32) {
    V16U a, b;
    a.q[0] = *(const v4u*)(Arow + k0 + g * 8);        // K = g*8 + 0..7
    a.q[1] = *(const v4u*)(Arow + k0 + 16 + g * 8);   // K = 16 + g*8 + 0..7
    b.q[0] = *(const v4u*)(Wrow + k0 + g * 8);        // K = g*8 + 0..7
    b.q[1] = *(const v4u*)(Wrow + k0 + 16 + g * 8);   // K = 16 + g*8 + 0..7
    acc = __builtin_amdgcn_wmma_f32_16x16x32_bf16(false, a.v, false, b.v,
                                                  (short)0, acc, false, false);
  }
  return acc;
}

// ---------------------------------------------------------------------------
// Conversion kernels
// ---------------------------------------------------------------------------
__global__ void k_cvt(const float* __restrict__ src, bf16_t* __restrict__ dst, int n) {
  int i = blockIdx.x * blockDim.x + threadIdx.x;
  if (i < n) dst[i] = f2bf(src[i]);
}
__global__ void k_cvtT(const float* __restrict__ src, bf16_t* __restrict__ dst,
                       int rows, int cols) {
  int i = blockIdx.x * blockDim.x + threadIdx.x;
  if (i < rows * cols) {
    int r = i / cols, c = i % cols;
    dst[(size_t)c * rows + r] = f2bf(src[i]);
  }
}
__global__ void k_initcnt(const float* __restrict__ src, float* __restrict__ dst, int n) {
  int i = blockIdx.x * blockDim.x + threadIdx.x;
  if (i < n) dst[i] = src[i];
}

// ---------------------------------------------------------------------------
// Generic small GEMM: out = act(A[M,K](bf16) * Wt[N,K](bf16)^T-storage + bias)
// One 16x16 tile per wave.  ACT: 0=none 1=relu 2=tanh
// ---------------------------------------------------------------------------
template <int ACT>
__global__ void k_gemm(const bf16_t* __restrict__ A, int lda,
                       const bf16_t* __restrict__ Wt, int ldw,
                       const float* __restrict__ bias,
                       float* __restrict__ outF, int ldoF,
                       bf16_t* __restrict__ outB, int ldoB,
                       int Mr, int Nc, int K) {
  int wave = blockIdx.x * (blockDim.x >> 5) + (threadIdx.x >> 5);
  int ctn = Nc >> 4;
  int ntiles = (Mr >> 4) * ctn;
  if (wave >= ntiles) return;                     // wave-uniform
  int rt = wave / ctn, ct = wave % ctn;
  int lane = threadIdx.x & 31, g = lane >> 4, ml = lane & 15;
  v8f acc = {};
  acc = bf16_mma_loop(A + (size_t)(rt * 16 + ml) * lda,
                      Wt + (size_t)(ct * 16 + ml) * ldw, K, g, acc);
  int col = ct * 16 + ml;
  float bv = bias ? bias[col] : 0.0f;
#pragma unroll
  for (int r = 0; r < 8; r++) {
    int row = rt * 16 + r + 8 * g;
    float v = acc[r] + bv;
    if (ACT == 1) v = fmaxf(v, 0.0f);
    if (ACT == 2) v = tanhf(v);
    if (outF) outF[(size_t)row * ldoF + col] = v;
    if (outB) outB[(size_t)row * ldoB + col] = f2bf(v);
  }
}

// ---------------------------------------------------------------------------
// LayerNorm (row of 512) -> patterns f32 (output), bf16 into pair buffer
// (ld 1024), and normalized qn bf16.
// ---------------------------------------------------------------------------
__global__ void k_ln(const float* __restrict__ pre,
                     const float* __restrict__ gamma, const float* __restrict__ beta,
                     float* __restrict__ outF, bf16_t* __restrict__ pairB,
                     bf16_t* __restrict__ qnB) {
  int row = blockIdx.x, tid = threadIdx.x;        // 256 threads, 512 cols
  __shared__ float red[256];
  __shared__ float sh[2];
  float a0 = pre[row * 512 + tid], a1 = pre[row * 512 + 256 + tid];
  red[tid] = a0 + a1;
  __syncthreads();
  for (int s = 128; s > 0; s >>= 1) { if (tid < s) red[tid] += red[tid + s]; __syncthreads(); }
  if (tid == 0) sh[0] = red[0] * (1.0f / 512.0f);
  __syncthreads();
  float mu = sh[0];
  float d0 = a0 - mu, d1 = a1 - mu;
  red[tid] = d0 * d0 + d1 * d1;
  __syncthreads();
  for (int s = 128; s > 0; s >>= 1) { if (tid < s) red[tid] += red[tid + s]; __syncthreads(); }
  if (tid == 0) sh[1] = rsqrtf(red[0] * (1.0f / 512.0f) + 1e-5f);
  __syncthreads();
  float rs = sh[1];
  float p0 = d0 * rs * gamma[tid] + beta[tid];
  float p1 = d1 * rs * gamma[256 + tid] + beta[256 + tid];
  outF[row * 512 + tid] = p0;
  outF[row * 512 + 256 + tid] = p1;
  pairB[row * 1024 + tid] = f2bf(p0);
  pairB[row * 1024 + 256 + tid] = f2bf(p1);
  red[tid] = p0 * p0 + p1 * p1;
  __syncthreads();
  for (int s = 128; s > 0; s >>= 1) { if (tid < s) red[tid] += red[tid + s]; __syncthreads(); }
  if (tid == 0) sh[0] = rsqrtf(red[0] + 1e-8f);
  __syncthreads();
  float rn = sh[0];
  qnB[row * 512 + tid] = f2bf(p0 * rn);
  qnB[row * 512 + 256 + tid] = f2bf(p1 * rn);
}

// ---------------------------------------------------------------------------
// Flash-attention split-K partial kernel with double-buffered async staging.
// grid = (NSPLIT, NH/HPW), block = 256 (8 waves).
// HD = head dim (64 or 32), HPW = heads per workgroup.
// ---------------------------------------------------------------------------
template <int HD, int HPW>
__global__ void k_flash(const float* __restrict__ mem, int Mtot, int rps,
                        const bf16_t* __restrict__ Q,     // [64, NH*HD]
                        const bf16_t* __restrict__ WkT,   // [NH*HD, 512]
                        const bf16_t* __restrict__ WvT,   // [NH*HD, 512]
                        float* __restrict__ pm, float* __restrict__ pl,
                        float* __restrict__ pacc) {
  extern __shared__ char smem[];
  float*  s_f32 = (float*)smem;                   // [CH][512] async staging
  bf16_t* s_mem = (bf16_t*)(s_f32 + CH * 512);    // [CH][512]
  bf16_t* s_k   = s_mem + CH * 512;               // [CH][HD]
  bf16_t* s_vT  = s_k + CH * HD;                  // [HD][CH]
  bf16_t* s_P   = s_vT + HD * CH;                 // [BQ][CH]
  float*  s_S   = (float*)(s_P + BQ * CH);        // [BQ][CH]
  float*  s_acc = s_S + BQ * CH;                  // [HPW][BQ][HD]
  float*  s_m   = s_acc + HPW * BQ * HD;          // [HPW][BQ]
  float*  s_l   = s_m + HPW * BQ;                 // [HPW][BQ]
  float*  s_al  = s_l + HPW * BQ;                 // [BQ]

  int tid = threadIdx.x, lane = tid & 31, wv = tid >> 5;
  int g = lane >> 4, ml = lane & 15;
  int split = blockIdx.x, hb = blockIdx.y * HPW;
  int start = split * rps;
  int end = dmin(Mtot, start + rps);
  constexpr int CT = HD / 16;                     // col tiles per head
  unsigned lbase = lds_addr(s_f32);

  // issue async global->LDS copy of one CH-row chunk (row clamped for tails)
  auto issue_chunk = [&](int cbase) {
    for (int seg = tid; seg < CH * 128; seg += 256) {   // 16B segments
      int r = seg >> 7, c4 = seg & 127;
      int rg = dmin(cbase + r, Mtot - 1);
      async_ld16(lbase + (unsigned)seg * 16u, mem + (size_t)rg * 512 + c4 * 4);
    }
  };

  for (int i = tid; i < HPW * BQ; i += 256) { s_m[i] = -3.0e38f; s_l[i] = 0.0f; }
  for (int i = tid; i < HPW * BQ * HD; i += 256) s_acc[i] = 0.0f;

  issue_chunk(start);
  wait_async0();
  __syncthreads();

  for (int base = start; base < end; base += CH) {
    int valid = dmin(CH, end - base);
    // convert staged fp32 chunk -> bf16 (zero padded rows)
    for (int i = tid; i < CH * 512; i += 256) {
      int r = i >> 9;
      float v = (r < valid) ? s_f32[i] : 0.0f;
      s_mem[i] = f2bf(v);
    }
    __syncthreads();
    // prefetch next chunk while we compute on this one
    if (base + CH < end) issue_chunk(base + CH);

    for (int hh = 0; hh < HPW; hh++) {
      int head = hb + hh;
      // --- K and V projections: memc[CH,512] @ W[:,head cols] (WMMA) ---
      for (int job = wv; job < 4 * CT; job += 8) {
        int which = job / (2 * CT);                      // 0=K 1=V
        int t = job % (2 * CT);
        int rt = t / CT, ct = t % CT;
        const bf16_t* WT = (which ? WvT : WkT) + (size_t)(head * HD + ct * 16 + ml) * 512;
        v8f a = {};
        a = bf16_mma_loop(s_mem + (rt * 16 + ml) * 512, WT, 512, g, a);
        int d = ct * 16 + ml;
#pragma unroll
        for (int r = 0; r < 8; r++) {
          int j = rt * 16 + r + 8 * g;
          if (which == 0) s_k[j * HD + d] = f2bf(a[r]);
          else            s_vT[d * CH + j] = f2bf(a[r]);  // store V transposed
        }
      }
      __syncthreads();
      // --- S = Q_head[64,HD] @ Kc^T -> [64,CH] (Kc row-major == Wt format) ---
      {
        int rt = wv >> 1, ct = wv & 1;                   // 8 tiles, 8 waves
        v8f a = {};
        a = bf16_mma_loop(Q + (size_t)(rt * 16 + ml) * (NHD * HD) + head * HD,
                          s_k + (ct * 16 + ml) * HD, HD, g, a);
        int col = ct * 16 + ml;
#pragma unroll
        for (int r = 0; r < 8; r++) s_S[(rt * 16 + r + 8 * g) * CH + col] = a[r];
      }
      __syncthreads();
      // --- online softmax (one thread per query) ---
      if (tid < BQ) {
        int q = tid;
        float mx = -3.0e38f;
        for (int j = 0; j < valid; j++) mx = fmaxf(mx, s_S[q * CH + j] * SCALE_F);
        float mo = s_m[hh * BQ + q];
        float mn2 = fmaxf(mo, mx);
        float al = __expf(mo - mn2);
        float sum = 0.0f;
        for (int j = 0; j < CH; j++) {
          float p = (j < valid) ? __expf(s_S[q * CH + j] * SCALE_F - mn2) : 0.0f;
          s_P[q * CH + j] = f2bf(p);
          sum += p;
        }
        s_m[hh * BQ + q] = mn2;
        s_l[hh * BQ + q] = s_l[hh * BQ + q] * al + sum;
        s_al[q] = al;
      }
      __syncthreads();
      // --- rescale acc by alpha ---
      for (int i = tid; i < BQ * HD; i += 256) s_acc[hh * BQ * HD + i] *= s_al[i / HD];
      __syncthreads();
      // --- acc += P[64,CH] @ Vc[CH,HD] (VcT is the Wt format) ---
      for (int t = wv; t < 4 * CT; t += 8) {
        int rt = t / CT, ct = t % CT;
        V16U a, b;
        const bf16_t* Arow = s_P + (rt * 16 + ml) * CH;
        const bf16_t* WT = s_vT + (ct * 16 + ml) * CH;
        a.q[0] = *(const v4u*)(Arow + g * 8);
        a.q[1] = *(const v4u*)(Arow + 16 + g * 8);
        b.q[0] = *(const v4u*)(WT + g * 8);
        b.q[1] = *(const v4u*)(WT + 16 + g * 8);
        int col = ct * 16 + ml;
        v8f c;
#pragma unroll
        for (int r = 0; r < 8; r++)
          c[r] = s_acc[hh * BQ * HD + (rt * 16 + r + 8 * g) * HD + col];
        c = __builtin_amdgcn_wmma_f32_16x16x32_bf16(false, a.v, false, b.v,
                                                    (short)0, c, false, false);
#pragma unroll
        for (int r = 0; r < 8; r++)
          s_acc[hh * BQ * HD + (rt * 16 + r + 8 * g) * HD + col] = c[r];
      }
      __syncthreads();
    } // heads

    wait_async0();          // next chunk landed in s_f32
    __syncthreads();
  }   // chunks

  for (int hh = 0; hh < HPW; hh++) {
    int head = hb + hh;
    for (int i = tid; i < BQ; i += 256) {
      pm[((size_t)split * NHD + head) * BQ + i] = s_m[hh * BQ + i];
      pl[((size_t)split * NHD + head) * BQ + i] = s_l[hh * BQ + i];
    }
    for (int i = tid; i < BQ * HD; i += 256)
      pacc[((size_t)split * NHD + head) * BQ * HD + i] = s_acc[hh * BQ * HD + i];
  }
}

// Combine split partials -> attn output [64, NH*HD] bf16.
template <int HD>
__global__ void k_combine(const float* __restrict__ pm, const float* __restrict__ pl,
                          const float* __restrict__ pacc, int nsplit,
                          bf16_t* __restrict__ outB) {
  int h = blockIdx.x, q = blockIdx.y, d = threadIdx.x;   // grid (NH,64), block HD
  float mstar = -3.0e38f;
  for (int s = 0; s < nsplit; s++)
    mstar = fmaxf(mstar, pm[((size_t)s * NHD + h) * BQ + q]);
  float L = 0.0f, o = 0.0f;
  for (int s = 0; s < nsplit; s++) {
    float w = __expf(pm[((size_t)s * NHD + h) * BQ + q] - mstar);
    L += pl[((size_t)s * NHD + h) * BQ + q] * w;
    o += pacc[(((size_t)s * NHD + h) * BQ + q) * HD + d] * w;
  }
  outB[(size_t)q * (NHD * HD) + h * HD + d] = f2bf(o / L);
}

// ---------------------------------------------------------------------------
// Cosine-sim + per-chunk top-5, async-staged fp32 tile + WMMA sims tiles.
// ---------------------------------------------------------------------------
__global__ void k_sims(const float* __restrict__ mem, int Mtot,
                       const bf16_t* __restrict__ qn,
                       float* __restrict__ candS, int* __restrict__ candI) {
  extern __shared__ char smem[];
  float*  s_f32 = (float*)smem;                   // [32][512] async staging
  bf16_t* s_mn  = (bf16_t*)(s_f32 + 32 * 512);    // [32][512]
  float*  s_S   = (float*)(s_mn + 32 * 512);      // [64][32]
  float*  s_red = s_S + 64 * 32;                  // [32][8]
  float*  s_inv = s_red + 32 * 8;                 // [32]
  float*  s_ts  = s_inv + 32;                     // [64][5]
  int*    s_ti  = (int*)(s_ts + 64 * 5);          // [64][5]
  int tid = threadIdx.x, lane = tid & 31, wv = tid >> 5;
  int g = lane >> 4, ml = lane & 15;
  int base0 = blockIdx.x * SIMS_CHUNK;
  int nsb = dmin(SIMS_CHUNK / 32, (Mtot - base0 + 31) / 32);
  unsigned lbase = lds_addr(s_f32);

  auto issue_sb = [&](int sb) {
    int cbase = base0 + sb * 32;
    for (int seg = tid; seg < 32 * 128; seg += 256) {
      int r = seg >> 7, c4 = seg & 127;
      int rg = dmin(cbase + r, Mtot - 1);
      async_ld16(lbase + (unsigned)seg * 16u, mem + (size_t)rg * 512 + c4 * 4);
    }
  };

  if (tid < 64)
    for (int k = 0; k < 5; k++) { s_ts[tid * 5 + k] = -3.0e38f; s_ti[tid * 5 + k] = 0; }

  issue_sb(0);
  wait_async0();
  __syncthreads();

  for (int sb = 0; sb < nsb; sb++) {
    int base = base0 + sb * 32;
    int valid = dmin(32, Mtot - base);
    { // row sum-of-squares from staged tile (8 threads per row)
      int r = tid >> 3, sub = tid & 7;
      float s = 0.0f;
      if (r < valid) {
        const float* row = s_f32 + r * 512 + sub * 64;
        for (int c = 0; c < 64; c++) { float v = row[c]; s += v * v; }
      }
      s_red[r * 8 + sub] = s;
    }
    __syncthreads();
    if (tid < 32) {
      float s = 0.0f;
      for (int k = 0; k < 8; k++) s += s_red[tid * 8 + k];
      s_inv[tid] = rsqrtf(s + 1e-8f);
    }
    __syncthreads();
    for (int i = tid; i < 32 * 512; i += 256) {
      int r = i >> 9;
      float v = (r < valid) ? s_f32[i] * s_inv[r] : 0.0f;
      s_mn[i] = f2bf(v);
    }
    __syncthreads();
    // prefetch next sub-block while WMMA runs
    if (sb + 1 < nsb) issue_sb(sb + 1);
    { // sims tile: qn[64,512] @ mn^T -> [64,32]
      int rt = wv >> 1, ct = wv & 1;
      v8f a = {};
      a = bf16_mma_loop(qn + (size_t)(rt * 16 + ml) * 512,
                        s_mn + (ct * 16 + ml) * 512, 512, g, a);
      int col = ct * 16 + ml;
#pragma unroll
      for (int r = 0; r < 8; r++) s_S[(rt * 16 + r + 8 * g) * 32 + col] = a[r];
    }
    __syncthreads();
    if (tid < 64) {
      int q = tid;
      for (int j = 0; j < valid; j++) {
        float s = s_S[q * 32 + j];
        if (s > s_ts[q * 5 + 4]) {
          int p = 4;
          while (p > 0 && s_ts[q * 5 + p - 1] < s) {
            s_ts[q * 5 + p] = s_ts[q * 5 + p - 1];
            s_ti[q * 5 + p] = s_ti[q * 5 + p - 1];
            p--;
          }
          s_ts[q * 5 + p] = s;
          s_ti[q * 5 + p] = base + j;
        }
      }
    }
    wait_async0();
    __syncthreads();
  }
  if (tid < 64)
    for (int k = 0; k < 5; k++) {
      candS[((size_t)blockIdx.x * 64 + tid) * 5 + k] = s_ts[tid * 5 + k];
      candI[((size_t)blockIdx.x * 64 + tid) * 5 + k] = s_ti[tid * 5 + k];
    }
}

__global__ void k_topk_final(const float* __restrict__ candS, const int* __restrict__ candI,
                             int nchunk, float* __restrict__ tops, float* __restrict__ counts) {
  int q = threadIdx.x;
  if (q >= 64) return;
  float ts[5]; int ti[5];
  for (int k = 0; k < 5; k++) { ts[k] = -3.0e38f; ti[k] = 0; }
  for (int c = 0; c < nchunk; c++)
    for (int k = 0; k < 5; k++) {
      float s = candS[((size_t)c * 64 + q) * 5 + k];
      int idx = candI[((size_t)c * 64 + q) * 5 + k];
      if (s > ts[4]) {
        int p = 4;
        while (p > 0 && ts[p - 1] < s) { ts[p] = ts[p - 1]; ti[p] = ti[p - 1]; p--; }
        ts[p] = s; ti[p] = idx;
      }
    }
  for (int k = 0; k < 5; k++) {
    tops[q * 5 + k] = ts[k];
    atomicAdd(&counts[ti[k]], 1.0f);
  }
}

__global__ void k_scorer(const bf16_t* __restrict__ s1, const float* __restrict__ Ws2,
                         const float* __restrict__ bs2, float* __restrict__ out) {
  int q = threadIdx.x;
  if (q >= 64) return;
  float acc = 0.0f;
  for (int k = 0; k < 256; k++) acc += bf2f(s1[q * 256 + k]) * Ws2[k];
  acc += bs2[0];
  out[q] = 1.0f / (1.0f + __expf(-acc));
}

// ---------------------------------------------------------------------------
// Host launcher
// ---------------------------------------------------------------------------
extern "C" void kernel_launch(void* const* d_in, const int* in_sizes, int n_in,
                              void* d_out, int out_size, void* d_ws, size_t ws_size,
                              hipStream_t stream) {
  (void)n_in; (void)out_size; (void)ws_size;
  const float* emb   = (const float*)d_in[0];
  const float* mem   = (const float*)d_in[1];
  const float* acnt  = (const float*)d_in[2];
  const float* W_in  = (const float*)d_in[3];
  const float* b_in  = (const float*)d_in[4];
  const float* W_pat = (const float*)d_in[5];
  const float* b_pat = (const float*)d_in[6];
  const float* gamma = (const float*)d_in[7];
  const float* beta  = (const float*)d_in[8];
  const float* Wt1   = (const float*)d_in[9];
  const float* bt1   = (const float*)d_in[10];
  const float* Wt2   = (const float*)d_in[11];
  const float* bt2   = (const float*)d_in[12];
  const float* Wq0   = (const float*)d_in[13];
  const float* Wk0   = (const float*)d_in[14];
  const float* Wv0   = (const float*)d_in[15];
  const float* Wo0   = (const float*)d_in[16];
  const float* Wq1   = (const float*)d_in[17];
  const float* Wk1   = (const float*)d_in[18];
  const float* Wv1   = (const float*)d_in[19];
  const float* Wo1   = (const float*)d_in[20];
  const float* Wc1   = (const float*)d_in[21];
  const float* bc1   = (const float*)d_in[22];
  const float* Wc2   = (const float*)d_in[23];
  const float* bc2   = (const float*)d_in[24];
  const float* Ws1   = (const float*)d_in[25];
  const float* bs1   = (const float*)d_in[26];
  const float* Ws2   = (const float*)d_in[27];
  const float* bs2   = (const float*)d_in[28];

  const int Mtot = in_sizes[2];                      // 100000
  float* out = (float*)d_out;
  float* o_pat  = out;
  float* o_tr   = out + 64 * 512;
  float* o_ret  = out + 2 * 64 * 512;
  float* o_tops = out + 3 * 64 * 512;
  float* o_cons = o_tops + 64 * 5;
  float* o_cnt  = o_cons + 64;

  // workspace carve-out
  char* p = (char*)d_ws;
  auto alloc = [&](size_t bytes) { void* r = p; p += (bytes + 255) & ~(size_t)255; return r; };
  bf16_t* embB  = (bf16_t*)alloc(64 * 384 * 2);
  bf16_t* WinT  = (bf16_t*)alloc(384 * 256 * 2);
  bf16_t* WpatT = (bf16_t*)alloc(256 * 512 * 2);
  bf16_t* Wt1T  = (bf16_t*)alloc(512 * 256 * 2);
  bf16_t* Wt2T  = (bf16_t*)alloc(256 * 512 * 2);
  bf16_t* Wq0T  = (bf16_t*)alloc(512 * 512 * 2);
  bf16_t* Wk0T  = (bf16_t*)alloc(512 * 512 * 2);
  bf16_t* Wv0T  = (bf16_t*)alloc(512 * 512 * 2);
  bf16_t* Wo0T  = (bf16_t*)alloc(512 * 512 * 2);
  bf16_t* Wq1T  = (bf16_t*)alloc(512 * 256 * 2);
  bf16_t* Wk1T  = (bf16_t*)alloc(512 * 256 * 2);
  bf16_t* Wv1T  = (bf16_t*)alloc(512 * 256 * 2);
  bf16_t* Wo1T  = (bf16_t*)alloc(256 * 512 * 2);
  bf16_t* Wc1T  = (bf16_t*)alloc(1024 * 512 * 2);
  bf16_t* Wc2T  = (bf16_t*)alloc(512 * 512 * 2);
  bf16_t* Ws1T  = (bf16_t*)alloc(1024 * 256 * 2);
  bf16_t* hidB  = (bf16_t*)alloc(64 * 256 * 2);
  bf16_t* pairB = (bf16_t*)alloc(64 * 1024 * 2);   // [patterns | retrieved] bf16
  bf16_t* qnB   = (bf16_t*)alloc(64 * 512 * 2);
  bf16_t* Q0B   = (bf16_t*)alloc(64 * 512 * 2);
  bf16_t* Q1B   = (bf16_t*)alloc(64 * 256 * 2);
  bf16_t* at0B  = (bf16_t*)alloc(64 * 512 * 2);
  bf16_t* at1B  = (bf16_t*)alloc(64 * 256 * 2);
  bf16_t* catB  = (bf16_t*)alloc(64 * 1024 * 2);   // [r0 | r1] bf16
  bf16_t* c1B   = (bf16_t*)alloc(64 * 512 * 2);
  bf16_t* s1B   = (bf16_t*)alloc(64 * 256 * 2);
  bf16_t* t1B   = (bf16_t*)alloc(64 * 256 * 2);
  float* patPre = (float*)alloc(64 * 512 * 4);
  float* pm0 = (float*)alloc((size_t)NSPLIT * NHD * 64 * 4);
  float* pl0 = (float*)alloc((size_t)NSPLIT * NHD * 64 * 4);
  float* pa0 = (float*)alloc((size_t)NSPLIT * NHD * 64 * 64 * 4);
  float* pm1 = (float*)alloc((size_t)NSPLIT * NHD * 64 * 4);
  float* pl1 = (float*)alloc((size_t)NSPLIT * NHD * 64 * 4);
  float* pa1 = (float*)alloc((size_t)NSPLIT * NHD * 64 * 32 * 4);
  const int nchunk = (Mtot + SIMS_CHUNK - 1) / SIMS_CHUNK;
  float* candS = (float*)alloc((size_t)nchunk * 64 * 5 * 4);
  int*   candI = (int*)alloc((size_t)nchunk * 64 * 5 * 4);

  auto nb = [](int n, int t) { return (n + t - 1) / t; };

  // --- weight / input conversion ---
  k_cvt<<<nb(64 * 384, 256), 256, 0, stream>>>(emb, embB, 64 * 384);
  k_cvtT<<<nb(384 * 256, 256), 256, 0, stream>>>(W_in, WinT, 384, 256);
  k_cvtT<<<nb(256 * 512, 256), 256, 0, stream>>>(W_pat, WpatT, 256, 512);
  k_cvtT<<<nb(512 * 256, 256), 256, 0, stream>>>(Wt1, Wt1T, 512, 256);
  k_cvtT<<<nb(256 * 512, 256), 256, 0, stream>>>(Wt2, Wt2T, 256, 512);
  k_cvtT<<<nb(512 * 512, 256), 256, 0, stream>>>(Wq0, Wq0T, 512, 512);
  k_cvtT<<<nb(512 * 512, 256), 256, 0, stream>>>(Wk0, Wk0T, 512, 512);
  k_cvtT<<<nb(512 * 512, 256), 256, 0, stream>>>(Wv0, Wv0T, 512, 512);
  k_cvtT<<<nb(512 * 512, 256), 256, 0, stream>>>(Wo0, Wo0T, 512, 512);
  k_cvtT<<<nb(512 * 256, 256), 256, 0, stream>>>(Wq1, Wq1T, 512, 256);
  k_cvtT<<<nb(512 * 256, 256), 256, 0, stream>>>(Wk1, Wk1T, 512, 256);
  k_cvtT<<<nb(512 * 256, 256), 256, 0, stream>>>(Wv1, Wv1T, 512, 256);
  k_cvtT<<<nb(256 * 512, 256), 256, 0, stream>>>(Wo1, Wo1T, 256, 512);
  k_cvtT<<<nb(1024 * 512, 256), 256, 0, stream>>>(Wc1, Wc1T, 1024, 512);
  k_cvtT<<<nb(512 * 512, 256), 256, 0, stream>>>(Wc2, Wc2T, 512, 512);
  k_cvtT<<<nb(1024 * 256, 256), 256, 0, stream>>>(Ws1, Ws1T, 1024, 256);

  // --- encoder ---
  k_gemm<1><<<nb(4 * 16, 8), 256, 0, stream>>>(embB, 384, WinT, 384, b_in,
      nullptr, 0, hidB, 256, 64, 256, 384);
  k_gemm<0><<<nb(4 * 32, 8), 256, 0, stream>>>(hidB, 256, WpatT, 256, b_pat,
      patPre, 512, nullptr, 0, 64, 512, 256);
  k_ln<<<64, 256, 0, stream>>>(patPre, gamma, beta, o_pat, pairB, qnB);

  // --- Q projections ---
  k_gemm<0><<<nb(4 * 32, 8), 256, 0, stream>>>(pairB, 1024, Wq0T, 512, nullptr,
      nullptr, 0, Q0B, 512, 64, 512, 512);
  k_gemm<0><<<nb(4 * 16, 8), 256, 0, stream>>>(pairB, 1024, Wq1T, 512, nullptr,
      nullptr, 0, Q1B, 256, 64, 256, 512);

  // --- traces ---
  k_gemm<1><<<nb(4 * 16, 8), 256, 0, stream>>>(pairB, 1024, Wt1T, 512, bt1,
      nullptr, 0, t1B, 256, 64, 256, 512);
  k_gemm<2><<<nb(4 * 32, 8), 256, 0, stream>>>(t1B, 256, Wt2T, 256, bt2,
      o_tr, 512, nullptr, 0, 64, 512, 256);

  // --- flash attention over memory bank (async-staged, double buffered) ---
  int rps = (Mtot + NSPLIT - 1) / NSPLIT;
  size_t shm0 = (size_t)CH * 512 * 4 + (size_t)CH * 512 * 2 + CH * 64 * 2 +
                64 * CH * 2 + BQ * CH * 2 + (size_t)BQ * CH * 4 +
                (size_t)4 * BQ * 64 * 4 + 4 * BQ * 4 * 2 + BQ * 4;
  size_t shm1 = (size_t)CH * 512 * 4 + (size_t)CH * 512 * 2 + CH * 32 * 2 +
                32 * CH * 2 + BQ * CH * 2 + (size_t)BQ * CH * 4 +
                (size_t)8 * BQ * 32 * 4 + 8 * BQ * 4 * 2 + BQ * 4;
  size_t shmS = (size_t)32 * 512 * 4 + 32 * 512 * 2 + 64 * 32 * 4 +
                32 * 8 * 4 + 32 * 4 + 64 * 5 * 4 + 64 * 5 * 4;
  hipFuncSetAttribute(reinterpret_cast<const void*>(&k_flash<64, 4>),
                      hipFuncAttributeMaxDynamicSharedMemorySize, (int)shm0);
  hipFuncSetAttribute(reinterpret_cast<const void*>(&k_flash<32, 8>),
                      hipFuncAttributeMaxDynamicSharedMemorySize, (int)shm1);
  hipFuncSetAttribute(reinterpret_cast<const void*>(&k_sims),
                      hipFuncAttributeMaxDynamicSharedMemorySize, (int)shmS);
  k_flash<64, 4><<<dim3(NSPLIT, 2), 256, shm0, stream>>>(mem, Mtot, rps, Q0B,
      Wk0T, Wv0T, pm0, pl0, pa0);
  k_flash<32, 8><<<dim3(NSPLIT, 1), 256, shm1, stream>>>(mem, Mtot, rps, Q1B,
      Wk1T, Wv1T, pm1, pl1, pa1);
  k_combine<64><<<dim3(NHD, 64), 64, 0, stream>>>(pm0, pl0, pa0, NSPLIT, at0B);
  k_combine<32><<<dim3(NHD, 64), 32, 0, stream>>>(pm1, pl1, pa1, NSPLIT, at1B);

  // --- output projections + consolidation MLP ---
  k_gemm<0><<<nb(4 * 32, 8), 256, 0, stream>>>(at0B, 512, Wo0T, 512, nullptr,
      nullptr, 0, catB, 1024, 64, 512, 512);
  k_gemm<0><<<nb(4 * 32, 8), 256, 0, stream>>>(at1B, 256, Wo1T, 256, nullptr,
      nullptr, 0, catB + 512, 1024, 64, 512, 256);
  k_gemm<1><<<nb(4 * 32, 8), 256, 0, stream>>>(catB, 1024, Wc1T, 1024, bc1,
      nullptr, 0, c1B, 512, 64, 512, 1024);
  k_gemm<0><<<nb(4 * 32, 8), 256, 0, stream>>>(c1B, 512, Wc2T, 512, bc2,
      o_ret, 512, pairB + 512, 1024, 64, 512, 512);

  // --- consolidation scorer ---
  k_gemm<1><<<nb(4 * 16, 8), 256, 0, stream>>>(pairB, 1024, Ws1T, 1024, bs1,
      nullptr, 0, s1B, 256, 64, 256, 1024);
  k_scorer<<<1, 64, 0, stream>>>(s1B, Ws2, bs2, o_cons);

  // --- top-k cosine similarity + counts ---
  k_initcnt<<<nb(Mtot, 256), 256, 0, stream>>>(acnt, o_cnt, Mtot);
  k_sims<<<nchunk, 256, shmS, stream>>>(mem, Mtot, qnB, candS, candI);
  k_topk_final<<<1, 64, 0, stream>>>(candS, candI, nchunk, o_tops, o_cnt);
}
